// Attention_12429635355261
// MI455X (gfx1250) — matrix operand order repaired and checked by reading, not compiled, not run on hardware
//
#include <hip/hip_runtime.h>

// ---------------------------------------------------------------------------
// Types
// ---------------------------------------------------------------------------
typedef _Float16 half_t;
typedef __attribute__((ext_vector_type(16))) _Float16 v16h;
typedef __attribute__((ext_vector_type(8)))  _Float16 v8h;
typedef __attribute__((ext_vector_type(8)))  float    v8f;

#define TM 64
#define TN 64
#define TK 32
#define LDSW 40   // halves per LDS row (32 + 8 pad) -> 80B stride, 16B aligned

// ---------------------------------------------------------------------------
// Generic WMMA GEMM: C[b][m][n] = sum_k A[b][m][k] * Bt[b][n][k]
// A fp16 [M x K] row-major (lda), Bt fp16 [N x K] row-major (ldb).
// Block = 128 threads (4 waves), 64x64 tile, each wave owns a 32x32 block
// (2x2 v_wmma_f32_16x16x32_f16 accumulators). Epilogue = functor.
// ---------------------------------------------------------------------------
template <typename Ep>
__global__ __launch_bounds__(128)
void gemm_wmma(const half_t* __restrict__ A, const half_t* __restrict__ Bt,
               int K, int lda, int ldb, long sA, long sB, Ep ep) {
    __shared__ half_t As[TM * LDSW];
    __shared__ half_t Bs[TN * LDSW];

    const int tid   = threadIdx.x;
    const int wave  = tid >> 5;
    const int lane  = tid & 31;
    const int hfl   = lane >> 4;   // 0/1: lane half
    const int idx   = lane & 15;
    const int waveM = wave >> 1;
    const int waveN = wave & 1;
    const int tileM = blockIdx.y * TM;
    const int tileN = blockIdx.x * TN;
    const int batch = blockIdx.z;

    const half_t* Ab = A  + (long)batch * sA;
    const half_t* Bb = Bt + (long)batch * sB;

    const v8f zero = {0.f, 0.f, 0.f, 0.f, 0.f, 0.f, 0.f, 0.f};
    v8f c[2][2];
    c[0][0] = zero; c[0][1] = zero; c[1][0] = zero; c[1][1] = zero;

    const int row = tid >> 1;        // 0..63 : tile row this thread fills
    const int seg = (tid & 1) * 16;  // 0 / 16 halves

    for (int k0 = 0; k0 < K; k0 += TK) {
        __syncthreads();
        {   // stage A tile (64 x 32 halves)
            const half_t* src = Ab + (long)(tileM + row) * lda + k0 + seg;
            v8h x0 = *(const v8h*)(src);
            v8h x1 = *(const v8h*)(src + 8);
            *(v8h*)&As[row * LDSW + seg]     = x0;
            *(v8h*)&As[row * LDSW + seg + 8] = x1;
            if (k0 + TK < K) __builtin_prefetch(src + TK, 0, 1);
        }
        {   // stage Bt tile (64 x 32 halves)
            const half_t* src = Bb + (long)(tileN + row) * ldb + k0 + seg;
            v8h x0 = *(const v8h*)(src);
            v8h x1 = *(const v8h*)(src + 8);
            *(v8h*)&Bs[row * LDSW + seg]     = x0;
            *(v8h*)&Bs[row * LDSW + seg + 8] = x1;
            if (k0 + TK < K) __builtin_prefetch(src + TK, 0, 1);
        }
        __syncthreads();

        // A fragments: 16x32, lane half selects contiguous K block h*16..h*16+15
        v16h a[2], b[2];
#pragma unroll
        for (int mi = 0; mi < 2; ++mi) {
            const half_t* p = &As[(waveM * 32 + mi * 16 + idx) * LDSW];
            v8h lo = *(const v8h*)(p + hfl * 16);
            v8h hi = *(const v8h*)(p + hfl * 16 + 8);
#pragma unroll
            for (int i = 0; i < 8; ++i) { a[mi][i] = lo[i]; a[mi][8 + i] = hi[i]; }
        }
        // B fragments: 32x16, lane half selects contiguous K block h*16..h*16+15
#pragma unroll
        for (int ni = 0; ni < 2; ++ni) {
            const half_t* p = &Bs[(waveN * 32 + ni * 16 + idx) * LDSW + hfl * 16];
            v8h lo = *(const v8h*)(p);
            v8h hi = *(const v8h*)(p + 8);
#pragma unroll
            for (int i = 0; i < 8; ++i) { b[ni][i] = lo[i]; b[ni][8 + i] = hi[i]; }
        }

#pragma unroll
        for (int mi = 0; mi < 2; ++mi)
#pragma unroll
            for (int ni = 0; ni < 2; ++ni)
                c[mi][ni] = __builtin_amdgcn_wmma_f32_16x16x32_f16(
                    false, a[mi], false, b[ni], (short)0, c[mi][ni], false, false);
    }

    // C layout: elem r of lane -> M = r + hfl*8, N = idx (per 16x16 tile)
#pragma unroll
    for (int mi = 0; mi < 2; ++mi)
#pragma unroll
        for (int ni = 0; ni < 2; ++ni)
#pragma unroll
            for (int r = 0; r < 8; ++r) {
                int m = tileM + waveM * 32 + mi * 16 + hfl * 8 + r;
                int n = tileN + waveN * 32 + ni * 16 + idx;
                ep(batch, m, n, c[mi][ni][r]);
            }
}

// ---------------------------------------------------------------------------
// Epilogue functors
// ---------------------------------------------------------------------------
struct EpQKV {  // qkv GEMM: n in [0,1536) -> q/k/v, heads of 64
    half_t *q, *k, *vt;
    __device__ void operator()(int b, int m, int n, float v) const {
        half_t hv = (half_t)v;
        int sec = n >> 9, r = n & 511, h = r >> 6, d = r & 63;
        long bh = (long)b * 8 + h;
        if (sec == 0)      q [(bh * 256 + m) * 64 + d] = hv;
        else if (sec == 1) k [(bh * 256 + m) * 64 + d] = hv;
        else               vt[(bh * 64 + d) * 256 + m] = hv;   // pre-transposed V
    }
};
struct EpStoreF32 {
    float* C; long sC; int ldc;
    __device__ void operator()(int b, int m, int n, float v) const {
        C[(long)b * sC + (long)m * ldc + n] = v;
    }
};
struct EpAttnV {  // batch = b*8+h ; pack heads back into [b][n][512]
    half_t* o;
    __device__ void operator()(int bh, int m, int n, float v) const {
        int b = bh >> 3, h = bh & 7;
        o[((long)b * 256 + m) * 512 + h * 64 + n] = (half_t)v;
    }
};
struct EpBiasF16 {  // out-proj: add bias, cast fp16
    half_t* o; const float* bias;
    __device__ void operator()(int b, int m, int n, float v) const {
        o[((long)b * 256 + m) * 512 + n] = (half_t)(v + bias[n]);
    }
};
struct EpSpecQK {  // n<256 -> qs, else ks
    half_t *qs, *ks;
    __device__ void operator()(int b, int m, int n, float v) const {
        half_t hv = (half_t)v;
        long base = ((long)b * 512 + m) * 256;
        if (n < 256) qs[base + n] = hv; else ks[base + n - 256] = hv;
    }
};
struct EpFinalF32 {
    float* o;
    __device__ void operator()(int b, int m, int n, float v) const {
        o[((long)b * 256 + m) * 512 + n] = v;
    }
};

// ---------------------------------------------------------------------------
// Elementwise / transpose / softmax / conv kernels (fp32 math)
// ---------------------------------------------------------------------------
__global__ void cvt16(const float* __restrict__ in, half_t* __restrict__ out, long n) {
    long i = (long)blockIdx.x * blockDim.x + threadIdx.x;
    if (i < n) out[i] = (half_t)in[i];
}

// wt[n][k] = w[k][n] (fp32 -> fp16), w row-major with leading dim ldw
__global__ void transpose_w(const float* __restrict__ w, half_t* __restrict__ wt,
                            int K, int Ncols, int ldw) {
    long i = (long)blockIdx.x * blockDim.x + threadIdx.x;
    if (i >= (long)K * Ncols) return;
    int k = (int)(i % K), n = (int)(i / K);
    wt[i] = (half_t)w[(long)k * ldw + n];
}

// xt16[b][d][t] = x[b][t][d]  (32 x 256 x 512 -> 32 x 512 x 256)
__global__ void transpose_x(const float* __restrict__ x, half_t* __restrict__ xt) {
    long i = (long)blockIdx.x * blockDim.x + threadIdx.x;
    int t = (int)(i & 255); long r = i >> 8;
    int d = (int)(r & 511); int b = (int)(r >> 9);
    xt[i] = (half_t)x[((long)b * 256 + t) * 512 + d];
}

// one block per row; softmax(row * scale), in place
__global__ void softmax_rows(float* __restrict__ data, int cols, float scale) {
    long row = blockIdx.x;
    float* p = data + row * (long)cols;
    __shared__ float red[256];
    float mx = -1e30f;
    for (int c = threadIdx.x; c < cols; c += 256) mx = fmaxf(mx, p[c] * scale);
    red[threadIdx.x] = mx; __syncthreads();
    for (int s = 128; s > 0; s >>= 1) {
        if (threadIdx.x < s) red[threadIdx.x] = fmaxf(red[threadIdx.x], red[threadIdx.x + s]);
        __syncthreads();
    }
    mx = red[0]; __syncthreads();
    float sum = 0.f;
    for (int c = threadIdx.x; c < cols; c += 256) {
        float e = __expf(p[c] * scale - mx); p[c] = e; sum += e;
    }
    red[threadIdx.x] = sum; __syncthreads();
    for (int s = 128; s > 0; s >>= 1) {
        if (threadIdx.x < s) red[threadIdx.x] += red[threadIdx.x + s];
        __syncthreads();
    }
    float inv = 1.0f / red[0];
    for (int c = threadIdx.x; c < cols; c += 256) p[c] *= inv;
}

// 3x3 SAME conv over score map, heads as channels: [32][8][256][256] -> fp16
__global__ void conv_spatial(const float* __restrict__ attn, const float* __restrict__ w,
                             const float* __restrict__ bias, half_t* __restrict__ out) {
    __shared__ float ws[8 * 8 * 9];
    __shared__ float bs[8];
    for (int i = threadIdx.x; i < 576; i += 256) ws[i] = w[i];
    if (threadIdx.x < 8) bs[threadIdx.x] = bias[threadIdx.x];
    __syncthreads();
    long gid = (long)blockIdx.x * 256 + threadIdx.x;   // ((b*8+o)*256+i)*256+j
    int j = (int)(gid & 255); long t = gid >> 8;
    int i = (int)(t & 255);   t >>= 8;
    int o = (int)(t & 7);     int b = (int)(t >> 3);
    float acc = bs[o];
    for (int c = 0; c < 8; ++c) {
        const float* base = attn + ((long)(b * 8 + c) << 16);
        const float* wk = &ws[(o * 8 + c) * 9];
#pragma unroll
        for (int di = 0; di < 3; ++di) {
            int ii = i + di - 1; if (ii < 0 || ii > 255) continue;
#pragma unroll
            for (int dj = 0; dj < 3; ++dj) {
                int jj = j + dj - 1; if (jj < 0 || jj > 255) continue;
                acc += base[(ii << 8) + jj] * wk[di * 3 + dj];
            }
        }
    }
    out[gid] = (half_t)acc;
}

// 3x3 SAME conv, single channel [32][512][512]; output TRANSPOSED fp16
// (outT[b][j][i] = conv(attn_s)[b][i][j]) so it feeds the final GEMM as Bt.
__global__ void conv_spectral(const float* __restrict__ s, const float* __restrict__ w,
                              const float* __restrict__ bias, half_t* __restrict__ outT) {
    long gid = (long)blockIdx.x * 256 + threadIdx.x;   // (b*512+i)*512+j
    int j = (int)(gid & 511); long t = gid >> 9;
    int i = (int)(t & 511);   int b = (int)(t >> 9);
    const float* base = s + ((long)b << 18);
    float acc = bias[0];
#pragma unroll
    for (int di = 0; di < 3; ++di) {
        int ii = i + di - 1; if (ii < 0 || ii > 511) continue;
#pragma unroll
        for (int dj = 0; dj < 3; ++dj) {
            int jj = j + dj - 1; if (jj < 0 || jj > 511) continue;
            acc += base[((long)ii << 9) + jj] * w[di * 3 + dj];
        }
    }
    outT[((long)b << 18) + ((long)j << 9) + i] = (half_t)acc;
}

// ---------------------------------------------------------------------------
// Host orchestration
// ---------------------------------------------------------------------------
extern "C" void kernel_launch(void* const* d_in, const int* in_sizes, int n_in,
                              void* d_out, int out_size, void* d_ws, size_t ws_size,
                              hipStream_t stream) {
    (void)in_sizes; (void)n_in; (void)out_size; (void)ws_size;
    const float* x        = (const float*)d_in[0];   // [32,256,512]
    const float* w_qkv    = (const float*)d_in[1];   // [512,1536]
    const float* w_out    = (const float*)d_in[2];   // [512,512]
    const float* b_out    = (const float*)d_in[3];   // [512]
    const float* w_sconv  = (const float*)d_in[4];   // [8,8,3,3]
    const float* b_sconv  = (const float*)d_in[5];   // [8]
    const float* w_spc    = (const float*)d_in[6];   // [1,1,3,3]
    const float* b_spc    = (const float*)d_in[7];   // [1]
    const float* w_qkv_sp = (const float*)d_in[8];   // [256,768]
    float* out = (float*)d_out;                      // [32,256,512]

    char* ws = (char*)d_ws;
    size_t off = 0;
    auto alloc = [&](size_t bytes) { size_t o = off; off += (bytes + 255) & ~(size_t)255; return o; };

    half_t* xh       = (half_t*)(ws + alloc(32L*256*512*2));    // fp16 x
    half_t* xt16     = (half_t*)(ws + alloc(32L*512*256*2));    // fp16 x^T
    half_t* wqkvt    = (half_t*)(ws + alloc(1536L*512*2));      // w_qkv^T
    half_t* woutt    = (half_t*)(ws + alloc(512L*512*2));       // w_out^T
    half_t* wspect   = (half_t*)(ws + alloc(512L*256*2));       // w_qkv_spec[:, :512]^T
    half_t* q16      = (half_t*)(ws + alloc(32L*8*256*64*2));   // [bh][n][d]
    half_t* k16      = (half_t*)(ws + alloc(32L*8*256*64*2));   // [bh][n][d]
    half_t* v16t     = (half_t*)(ws + alloc(32L*8*64*256*2));   // [bh][d][n]
    float*  dots     = (float*) (ws + alloc(32L*8*256*256*4));  // reused as dots_s (33.5MB<=67MB)
    half_t* attnc16  = (half_t*)(ws + alloc(32L*8*256*256*2));  // conv'd scores fp16
    half_t* out16    = (half_t*)(ws + alloc(32L*256*512*2));    // attn@v packed
    half_t* output16 = (half_t*)(ws + alloc(32L*256*512*2));    // out-proj fp16
    half_t* qs16     = (half_t*)(ws + alloc(32L*512*256*2));
    half_t* ks16     = (half_t*)(ws + alloc(32L*512*256*2));
    half_t* at16     = (half_t*)(ws + alloc(32L*512*512*2));    // attn_s^T fp16
    float*  dots_s   = dots;

    const float SCALE = 0.125f;  // 64^-0.5

    // ---- staging / transposes ----
    cvt16<<<16384, 256, 0, stream>>>(x, xh, 32L*256*512);
    transpose_w<<<(1536*512 + 255)/256, 256, 0, stream>>>(w_qkv,    wqkvt,  512, 1536, 1536);
    transpose_w<<<(512*512  + 255)/256, 256, 0, stream>>>(w_out,    woutt,  512, 512,  512);
    transpose_w<<<(256*512  + 255)/256, 256, 0, stream>>>(w_qkv_sp, wspect, 256, 512,  768);
    transpose_x<<<16384, 256, 0, stream>>>(x, xt16);

    // ---- spatial attention ----
    // qkv = x @ w_qkv  (scatter q/k/v, V pre-transposed)
    gemm_wmma<<<dim3(24, 4, 32), 128, 0, stream>>>(
        xh, wqkvt, 512, 512, 512, 131072L, 0L, EpQKV{q16, k16, v16t});
    // dots = q @ k^T
    gemm_wmma<<<dim3(4, 4, 256), 128, 0, stream>>>(
        q16, k16, 64, 64, 64, 16384L, 16384L, EpStoreF32{dots, 65536L, 256});
    softmax_rows<<<65536, 256, 0, stream>>>(dots, 256, SCALE);
    conv_spatial<<<65536, 256, 0, stream>>>(dots, w_sconv, b_sconv, attnc16);
    // out = attn @ v
    gemm_wmma<<<dim3(1, 4, 256), 128, 0, stream>>>(
        attnc16, v16t, 256, 256, 256, 65536L, 16384L, EpAttnV{out16});
    // output = out @ w_out + b_out
    gemm_wmma<<<dim3(8, 4, 32), 128, 0, stream>>>(
        out16, woutt, 512, 512, 512, 131072L, 0L, EpBiasF16{output16, b_out});

    // ---- spectral attention (vs is dead code in the reference) ----
    gemm_wmma<<<dim3(8, 8, 32), 128, 0, stream>>>(
        xt16, wspect, 256, 256, 256, 131072L, 0L, EpSpecQK{qs16, ks16});
    gemm_wmma<<<dim3(8, 8, 32), 128, 0, stream>>>(
        qs16, ks16, 256, 256, 256, 131072L, 131072L, EpStoreF32{dots_s, 262144L, 512});
    softmax_rows<<<16384, 256, 0, stream>>>(dots_s, 512, SCALE);
    conv_spectral<<<32768, 256, 0, stream>>>(dots_s, w_spc, b_spc, at16);

    // ---- final: output @ attn_s ----
    gemm_wmma<<<dim3(8, 4, 32), 128, 0, stream>>>(
        output16, at16, 512, 512, 512, 131072L, 262144L, EpFinalF32{out});
}
